// MoELayer_45904610459763
// MI455X (gfx1250) — matrix operand order repaired and checked by reading, not compiled, not run on hardware
//
#include <hip/hip_runtime.h>
#include <hip/hip_bf16.h>

#define B_ 8
#define N_ 2048
#define D_ 1024
#define E_ 8
#define H_ 4096
#define C_ 256   // N * CAPACITY_FACTOR / E

// LDS tile geometry: BM=64, BN=256, KT=32, padded row stride 40 bf16 (80B)
#define LDS_STRIDE 40
#define A_ELEMS (64 * LDS_STRIDE)    // 2560
#define B_ELEMS (256 * LDS_STRIDE)   // 10240
#define BUF_ELEMS (A_ELEMS + B_ELEMS) // 12800 bf16 per buffer

typedef __attribute__((ext_vector_type(8)))  __bf16 v8bf;
typedef __attribute__((ext_vector_type(16))) __bf16 v16bf;
typedef __attribute__((ext_vector_type(8)))  float  v8f;

union ABu { v8bf h[2]; v16bf f; };

__device__ inline float gelu_tanh(float x) {
    // jax.nn.gelu default (approximate=True)
    const float c0 = 0.7978845608028654f; // sqrt(2/pi)
    const float c1 = 0.044715f;
    float x3 = x * x * x;
    return 0.5f * x * (1.0f + tanhf(c0 * (x + c1 * x3)));
}

// ---------------- elementwise prep kernels ----------------

__global__ void zero_f32_kernel(float* __restrict__ p, long total) {
    long stride = (long)gridDim.x * blockDim.x;
    for (long i = (long)blockIdx.x * blockDim.x + threadIdx.x; i < total; i += stride)
        p[i] = 0.0f;
}

__global__ void zero_bf16_kernel(__bf16* __restrict__ p, long total) {
    long stride = (long)gridDim.x * blockDim.x;
    for (long i = (long)blockIdx.x * blockDim.x + threadIdx.x; i < total; i += stride)
        p[i] = (__bf16)0.0f;
}

__global__ void convert_x_kernel(const float* __restrict__ x, __bf16* __restrict__ xb, long total) {
    long stride = (long)gridDim.x * blockDim.x;
    for (long i = (long)blockIdx.x * blockDim.x + threadIdx.x; i < total; i += stride)
        xb[i] = (__bf16)x[i];
}

// W1 (E,D,H) fp32 -> W1t (E,H,D) bf16
__global__ void transpose_w1_kernel(const float* __restrict__ W1, __bf16* __restrict__ W1t, long total) {
    long stride = (long)gridDim.x * blockDim.x;
    for (long i = (long)blockIdx.x * blockDim.x + threadIdx.x; i < total; i += stride) {
        int d = (int)(i % D_);
        long t = i / D_;
        int h = (int)(t % H_);
        int e = (int)(t / H_);
        W1t[i] = (__bf16)W1[((long)e * D_ + d) * H_ + h];
    }
}

// W2 (E,H,D) fp32 -> W2t (E,D,H) bf16
__global__ void transpose_w2_kernel(const float* __restrict__ W2, __bf16* __restrict__ W2t, long total) {
    long stride = (long)gridDim.x * blockDim.x;
    for (long i = (long)blockIdx.x * blockDim.x + threadIdx.x; i < total; i += stride) {
        int h = (int)(i % H_);
        long t = i / H_;
        int d = (int)(t % D_);
        int e = (int)(t / D_);
        W2t[i] = (__bf16)W2[((long)e * H_ + h) * D_ + d];
    }
}

// ---------------- router ----------------

__global__ __launch_bounds__(256) void router_logits_kernel(
    const float* __restrict__ x, const float* __restrict__ Wg,
    int* __restrict__ eidx, float* __restrict__ gtok)
{
    const int lane = threadIdx.x & 31;
    const int wave = threadIdx.x >> 5;
    const int t = blockIdx.x * 8 + wave;          // token index in [0, B*N)
    const float* xr = x + (size_t)t * D_;

    float acc[E_];
    #pragma unroll
    for (int e = 0; e < E_; ++e) acc[e] = 0.0f;

    for (int d = lane; d < D_; d += 32) {
        float xv = xr[d];
        const float* wr = Wg + d * E_;
        #pragma unroll
        for (int e = 0; e < E_; ++e) acc[e] += xv * wr[e];
    }
    #pragma unroll
    for (int e = 0; e < E_; ++e)
        for (int off = 16; off > 0; off >>= 1)
            acc[e] += __shfl_xor(acc[e], off, 32);

    if (lane == 0) {
        float m = acc[0]; int arg = 0;
        #pragma unroll
        for (int e = 1; e < E_; ++e) if (acc[e] > m) { m = acc[e]; arg = e; }
        float s = 0.0f;
        #pragma unroll
        for (int e = 0; e < E_; ++e) s += expf(acc[e] - m);
        eidx[t] = arg;
        gtok[t] = 1.0f / s;   // exp(max-max)/sum
    }
}

__global__ void router_assign_kernel(
    const int* __restrict__ eidx, const float* __restrict__ gtok,
    int* __restrict__ assign, float* __restrict__ gslot)
{
    int tid = threadIdx.x;         // 0..63
    if (tid >= B_ * E_) return;
    int b = tid >> 3;
    int e = tid & 7;
    int base = (b * E_ + e) * C_;
    int cnt = 0;
    for (int n = 0; n < N_; ++n) {
        if (eidx[b * N_ + n] == e) {
            if (cnt < C_) {
                assign[base + cnt] = n;
                gslot[base + cnt] = gtok[b * N_ + n];
            }
            ++cnt;
        }
    }
    for (int c = (cnt < C_ ? cnt : C_); c < C_; ++c) {
        assign[base + c] = -1;
        gslot[base + c] = 0.0f;
    }
}

// ---------------- GEMM 1: h = gelu(gather(X) @ W1) ----------------
// grid (H/256, C/64, B*E), block 256 = 8 waves (4M x 2N), wave tile 16x128.
__global__ __launch_bounds__(256) void gemm1_kernel(
    const __bf16* __restrict__ Xbf, const int* __restrict__ assign,
    const __bf16* __restrict__ W1t, const __bf16* __restrict__ zrow,
    __bf16* __restrict__ hbuf)
{
    __shared__ __attribute__((aligned(16))) __bf16 lds[2 * BUF_ELEMS];

    const int tid  = threadIdx.x;
    const int lane = tid & 31;
    const int wave = tid >> 5;
    const int be = blockIdx.z, b = be >> 3, e = be & 7;
    const int m_blk = blockIdx.y * 64;
    const int n_blk = blockIdx.x * 256;

    // ---- global staging roles (all 256 threads) ----
    const int rowA = tid >> 2;           // 0..63
    const int chA  = tid & 3;            // 0..3 (16B chunks of a 32-elem row)
    const int token = assign[be * C_ + m_blk + rowA];
    const __bf16* aptr = (token >= 0 ? Xbf + ((size_t)b * N_ + token) * D_ : zrow) + chA * 8;
    const __bf16* bptr = W1t + ((size_t)e * H_ + n_blk + tid) * D_;   // col tid, K contiguous

    // ---- compute roles ----
    const int mw = wave >> 1, nw = wave & 1;
    const int lrow = lane & 15;
    const int kba = (lane < 16) ? 0 : 16;
    const int kbb = (lane < 16) ? 0 : 16;
    const int aoff = (mw * 16 + lrow) * LDS_STRIDE + kba;
    const int boff = (nw * 128 + lrow) * LDS_STRIDE + kbb;

    v8f acc[8];
    #pragma unroll
    for (int j = 0; j < 8; ++j)
        #pragma unroll
        for (int i = 0; i < 8; ++i) acc[j][i] = 0.0f;

    // prefetch k-step 0
    v8bf ra  = *(const v8bf*)(aptr);
    v8bf rb0 = *(const v8bf*)(bptr);
    v8bf rb1 = *(const v8bf*)(bptr + 8);
    v8bf rb2 = *(const v8bf*)(bptr + 16);
    v8bf rb3 = *(const v8bf*)(bptr + 24);

    const int NK = D_ / 32;   // 32
    for (int kt = 0; kt < NK; ++kt) {
        __bf16* Al = lds + (kt & 1) * BUF_ELEMS;
        __bf16* Bl = Al + A_ELEMS;
        // stage current regs -> LDS buffer kt&1
        *(v8bf*)(Al + rowA * LDS_STRIDE + chA * 8) = ra;
        *(v8bf*)(Bl + tid * LDS_STRIDE)      = rb0;
        *(v8bf*)(Bl + tid * LDS_STRIDE + 8)  = rb1;
        *(v8bf*)(Bl + tid * LDS_STRIDE + 16) = rb2;
        *(v8bf*)(Bl + tid * LDS_STRIDE + 24) = rb3;
        // issue next k-step's global loads (latency hidden behind compute)
        if (kt + 1 < NK) {
            int k0 = (kt + 1) * 32;
            ra  = *(const v8bf*)(aptr + k0);
            rb0 = *(const v8bf*)(bptr + k0);
            rb1 = *(const v8bf*)(bptr + k0 + 8);
            rb2 = *(const v8bf*)(bptr + k0 + 16);
            rb3 = *(const v8bf*)(bptr + k0 + 24);
        }
        __syncthreads();
        // compute from LDS buffer kt&1 with rolling B-fragment buffer
        ABu a, bcur, bnxt;
        a.h[0] = *(const v8bf*)(Al + aoff);
        a.h[1] = *(const v8bf*)(Al + aoff + 8);
        bcur.h[0] = *(const v8bf*)(Bl + boff);
        bcur.h[1] = *(const v8bf*)(Bl + boff + 8);
        #pragma unroll
        for (int j = 0; j < 8; ++j) {
            if (j < 7) {
                const __bf16* bp = Bl + boff + (j + 1) * (16 * LDS_STRIDE);
                bnxt.h[0] = *(const v8bf*)(bp);
                bnxt.h[1] = *(const v8bf*)(bp + 8);
            }
            acc[j] = __builtin_amdgcn_wmma_f32_16x16x32_bf16(
                false, a.f, false, bcur.f, (short)0, acc[j], false, false);
            bcur = bnxt;
        }
    }

    // epilogue: gelu, store bf16 h
    const int mbase = (lane < 16) ? 0 : 8;
    __bf16* hs = hbuf + ((size_t)be * C_ + m_blk + mw * 16) * H_;
    #pragma unroll
    for (int j = 0; j < 8; ++j) {
        int col = n_blk + nw * 128 + j * 16 + lrow;
        #pragma unroll
        for (int i = 0; i < 8; ++i)
            hs[(size_t)(mbase + i) * H_ + col] = (__bf16)gelu_tanh(acc[j][i]);
    }
}

// ---------------- GEMM 2: out[token] = gate * (h @ W2) ----------------
// grid (D/256, C/64, B*E), block 256 = 8 waves (4M x 2N).
__global__ __launch_bounds__(256) void gemm2_kernel(
    const __bf16* __restrict__ hbuf, const int* __restrict__ assign,
    const float* __restrict__ gslot, const __bf16* __restrict__ W2t,
    float* __restrict__ out)
{
    __shared__ __attribute__((aligned(16))) __bf16 lds[2 * BUF_ELEMS];

    const int tid  = threadIdx.x;
    const int lane = tid & 31;
    const int wave = tid >> 5;
    const int be = blockIdx.z, b = be >> 3, e = be & 7;
    const int m_blk = blockIdx.y * 64;
    const int n_blk = blockIdx.x * 256;

    const int rowA = tid >> 2;
    const int chA  = tid & 3;
    const __bf16* aptr = hbuf + ((size_t)be * C_ + m_blk + rowA) * H_ + chA * 8;
    const __bf16* bptr = W2t + ((size_t)e * D_ + n_blk + tid) * H_;

    const int mw = wave >> 1, nw = wave & 1;
    const int lrow = lane & 15;
    const int kba = (lane < 16) ? 0 : 16;
    const int kbb = (lane < 16) ? 0 : 16;
    const int aoff = (mw * 16 + lrow) * LDS_STRIDE + kba;
    const int boff = (nw * 128 + lrow) * LDS_STRIDE + kbb;

    v8f acc[8];
    #pragma unroll
    for (int j = 0; j < 8; ++j)
        #pragma unroll
        for (int i = 0; i < 8; ++i) acc[j][i] = 0.0f;

    v8bf ra  = *(const v8bf*)(aptr);
    v8bf rb0 = *(const v8bf*)(bptr);
    v8bf rb1 = *(const v8bf*)(bptr + 8);
    v8bf rb2 = *(const v8bf*)(bptr + 16);
    v8bf rb3 = *(const v8bf*)(bptr + 24);

    const int NK = H_ / 32;   // 128
    for (int kt = 0; kt < NK; ++kt) {
        __bf16* Al = lds + (kt & 1) * BUF_ELEMS;
        __bf16* Bl = Al + A_ELEMS;
        *(v8bf*)(Al + rowA * LDS_STRIDE + chA * 8) = ra;
        *(v8bf*)(Bl + tid * LDS_STRIDE)      = rb0;
        *(v8bf*)(Bl + tid * LDS_STRIDE + 8)  = rb1;
        *(v8bf*)(Bl + tid * LDS_STRIDE + 16) = rb2;
        *(v8bf*)(Bl + tid * LDS_STRIDE + 24) = rb3;
        if (kt + 1 < NK) {
            int k0 = (kt + 1) * 32;
            ra  = *(const v8bf*)(aptr + k0);
            rb0 = *(const v8bf*)(bptr + k0);
            rb1 = *(const v8bf*)(bptr + k0 + 8);
            rb2 = *(const v8bf*)(bptr + k0 + 16);
            rb3 = *(const v8bf*)(bptr + k0 + 24);
        }
        __syncthreads();
        ABu a, bcur, bnxt;
        a.h[0] = *(const v8bf*)(Al + aoff);
        a.h[1] = *(const v8bf*)(Al + aoff + 8);
        bcur.h[0] = *(const v8bf*)(Bl + boff);
        bcur.h[1] = *(const v8bf*)(Bl + boff + 8);
        #pragma unroll
        for (int j = 0; j < 8; ++j) {
            if (j < 7) {
                const __bf16* bp = Bl + boff + (j + 1) * (16 * LDS_STRIDE);
                bnxt.h[0] = *(const v8bf*)(bp);
                bnxt.h[1] = *(const v8bf*)(bp + 8);
            }
            acc[j] = __builtin_amdgcn_wmma_f32_16x16x32_bf16(
                false, a.f, false, bcur.f, (short)0, acc[j], false, false);
            bcur = bnxt;
        }
    }

    // epilogue: gate-scale and scatter to token rows (pre-zeroed output)
    const int mbase = (lane < 16) ? 0 : 8;
    #pragma unroll
    for (int i = 0; i < 8; ++i) {
        int c = m_blk + mw * 16 + mbase + i;
        int token = assign[be * C_ + c];
        if (token >= 0) {
            float g = gslot[be * C_ + c];
            float* orow = out + ((size_t)b * N_ + token) * D_;
            #pragma unroll
            for (int j = 0; j < 8; ++j)
                orow[n_blk + nw * 128 + j * 16 + lrow] = g * acc[j][i];
        }
    }
}

// ---------------- launch ----------------

extern "C" void kernel_launch(void* const* d_in, const int* in_sizes, int n_in,
                              void* d_out, int out_size, void* d_ws, size_t ws_size,
                              hipStream_t stream) {
    (void)in_sizes; (void)n_in; (void)out_size; (void)ws_size;
    const float* x  = (const float*)d_in[0];   // (B,N,D)
    const float* Wg = (const float*)d_in[1];   // (D,E)
    const float* W1 = (const float*)d_in[2];   // (E,D,H)
    const float* W2 = (const float*)d_in[3];   // (E,H,D)
    float* out = (float*)d_out;                // (B,N,D)

    char* ws = (char*)d_ws;
    size_t off = 0;
    auto take = [&](size_t bytes) -> char* {
        char* p = ws + off;
        off = (off + bytes + 255) & ~(size_t)255;
        return p;
    };
    __bf16* Xbf   = (__bf16*)take((size_t)B_ * N_ * D_ * 2);      //  32 MB
    __bf16* W1t   = (__bf16*)take((size_t)E_ * H_ * D_ * 2);      //  64 MB
    __bf16* W2t   = (__bf16*)take((size_t)E_ * D_ * H_ * 2);      //  64 MB
    __bf16* hbuf  = (__bf16*)take((size_t)B_ * E_ * C_ * H_ * 2); // 128 MB
    __bf16* zrow  = (__bf16*)take((size_t)D_ * 2);                // zero gather row
    int*    eidx  = (int*)  take((size_t)B_ * N_ * 4);
    float*  gtok  = (float*)take((size_t)B_ * N_ * 4);
    int*    assign= (int*)  take((size_t)B_ * E_ * C_ * 4);
    float*  gslot = (float*)take((size_t)B_ * E_ * C_ * 4);

    const long xtotal = (long)B_ * N_ * D_;
    const long wtotal = (long)E_ * D_ * H_;

    zero_f32_kernel<<<8192, 256, 0, stream>>>(out, xtotal);
    zero_bf16_kernel<<<1, 256, 0, stream>>>(zrow, (long)D_);
    convert_x_kernel<<<8192, 256, 0, stream>>>(x, Xbf, xtotal);
    transpose_w1_kernel<<<16384, 256, 0, stream>>>(W1, W1t, wtotal);
    transpose_w2_kernel<<<16384, 256, 0, stream>>>(W2, W2t, wtotal);

    router_logits_kernel<<<(B_ * N_) / 8, 256, 0, stream>>>(x, Wg, eidx, gtok);
    router_assign_kernel<<<1, 64, 0, stream>>>(eidx, gtok, assign, gslot);

    dim3 g1(H_ / 256, C_ / 64, B_ * E_);
    gemm1_kernel<<<g1, 256, 0, stream>>>(Xbf, assign, W1t, zrow, hbuf);

    dim3 g2(D_ / 256, C_ / 64, B_ * E_);
    gemm2_kernel<<<g2, 256, 0, stream>>>(hbuf, assign, gslot, W2t, out);
}
